// HyperRadialNeuralFourierCelularAutomata_47665547051215
// MI455X (gfx1250) — matrix operand order repaired and checked by reading, not compiled, not run on hardware
//
#include <hip/hip_runtime.h>

typedef _Float16 v16h __attribute__((ext_vector_type(16)));
typedef _Float16 v8h  __attribute__((ext_vector_type(8)));
typedef float    v8f  __attribute__((ext_vector_type(8)));

#define JJ    421
#define HDC   256
#define JD    (JJ*HDC)          // 107776
#define BATCH 8
#define RBF   64
#define NCAH  32
#define NSTEPS 20

// ---------------------------------------------------------------- scalars ---
__global__ void k_scalars(const float* __restrict__ din, const float* __restrict__ sin_,
                          const float* __restrict__ mi, const float* __restrict__ mo,
                          const float* __restrict__ cw, float* __restrict__ ws) {
  int t = threadIdx.x;
  if (t < BATCH) {
    float s = 0.f;
    for (int i = 0; i < 324; ++i) s += din[t*324 + i];
    for (int i = 0; i < 81;  ++i) s += sin_[t*81 + i];
    float pin = mi[t] * 0.01f, pout = mo[t] * 0.01f;
    for (int i = 0; i < 16; ++i) {
      float div = powf(10000.f, (float)(2*(i/2)) * (1.f/16.f));
      float ain = pin / div, aout = pout / div;
      float pe = ((i & 1) == 0) ? 0.5f*(sinf(ain)+cosf(ain))
                                : 0.5f*(cosf(ain)+sinf(aout));
      s += pe;
    }
    ws[t] = s;
  } else if (t < BATCH + 5) {
    int o = t - BATCH;
    float c = 0.f;
    for (int r = 0; r < RBF; ++r) c += cw[o*RBF + r];
    ws[8 + o] = c;
  }
}

// ------------------------------------------------------------------- init ---
// x0[b,o,jd] = tanh( A[o,jd] - 2 p B[o,jd] + p^2 C[o] + cb[o] ),  p = s_b*hdc
// Single pass over the 27.6MB rbf tensor; 431KB-strided rows get explicit
// global_prefetch_b8 (speculative, safe to over-run).
__global__ void k_init(const float* __restrict__ rbf, const float* __restrict__ hdc,
                       const float* __restrict__ cw, const float* __restrict__ cb,
                       const float* __restrict__ sc, float* __restrict__ x0) {
  __shared__ float w[5*RBF];
  __shared__ float sS[BATCH], sC[5], sCB[5];
  int t = threadIdx.x;
  for (int i = t; i < 5*RBF; i += 256) w[i] = cw[i];
  if (t < BATCH) sS[t] = sc[t];
  if (t < 5) { sC[t] = sc[8 + t]; sCB[t] = cb[t]; }
  __syncthreads();
  int jd = blockIdx.x * 256 + t;
  if (jd >= JD) return;
  float aA[5] = {0,0,0,0,0}, aB[5] = {0,0,0,0,0};
  for (int r = 0; r < RBF; ++r) {
    int rp = (r + 6 < RBF) ? r + 6 : RBF - 1;
    __builtin_prefetch(&rbf[(size_t)rp*JD + jd], 0, 0);   // global_prefetch_b8
    float v = rbf[(size_t)r*JD + jd];
    float v2 = v*v;
    #pragma unroll
    for (int o = 0; o < 5; ++o) { float ww = w[o*RBF + r]; aA[o] += ww*v2; aB[o] += ww*v; }
  }
  for (int b = 0; b < BATCH; ++b) {
    float p = sS[b] * hdc[(size_t)b*JD + jd];
    float p2 = p*p;
    #pragma unroll
    for (int o = 0; o < 5; ++o) {
      float v = aA[o] - 2.f*p*aB[o] + p2*sC[o] + sCB[o];
      x0[((size_t)b*5 + o)*JD + jd] = tanhf(v);
    }
  }
}

// --------------------------------------------------------------- NCA step ---
// One block per (b, j) row; 16 waves, each wave owns a 16-pixel N-tile.
// GEMM1: h[32,16] = W1aug[32,64] x im2col[64,16]  (4x wmma f32_16x16x32_f16)
//        K=45 is the bias row (w1p[:,45]=b1, icol[:,45]=1) -> h=relu(acc)
// GEMM2: dx[5,16] = W2pad[16,32] x h               (1x wmma)
__global__ __launch_bounds__(512) void k_nca(
    const float* __restrict__ xin, float* __restrict__ xout,
    const float* __restrict__ w1, const float* __restrict__ b1,
    const float* __restrict__ w2, const float* __restrict__ b2) {
  __shared__ __attribute__((aligned(32))) float    xtp[3*5*258]; // padded cols, 15480 B
  __shared__ __attribute__((aligned(32))) _Float16 icol[256*64]; // im2col, 32 KB
  __shared__ __attribute__((aligned(32))) _Float16 w1p[32*64];   // 4 KB
  __shared__ __attribute__((aligned(32))) _Float16 w2p[16*32];   // 1 KB
  __shared__ int soff[64];

  int b = blockIdx.x / JJ, j = blockIdx.x % JJ;
  int t = threadIdx.x, lane = t & 31, wv = t >> 5;
  int lh = lane >> 4, lm = lane & 15;

  const float* xb = xin + (size_t)b*5*JD;

  // ---- phase 1: padded x tile, packed weights, im2col source-offset table
  #pragma unroll
  for (int rr = 0; rr < 3; ++rr) {
    int jjr = j + rr - 1;
    bool vrow = (jjr >= 0) && (jjr < JJ);
    #pragma unroll
    for (int c = 0; c < 5; ++c) {
      if (t < 256)
        xtp[(rr*5 + c)*258 + 1 + t] = vrow ? xb[(size_t)c*JD + jjr*256 + t] : 0.f;
    }
  }
  if (t < 30) xtp[(t >> 1)*258 + ((t & 1) ? 257 : 0)] = 0.f;   // border columns

  for (int i = t; i < 32*64; i += 512) {
    int m = i >> 6, K = i & 63;
    float v = (K < 45) ? w1[m*45 + K] : ((K == 45) ? b1[m] : 0.f);
    w1p[i] = (_Float16)v;
  }
  {
    int m = t >> 5, K = t & 31;                                 // t covers 16x32
    w2p[t] = (m < 5) ? (_Float16)w2[m*32 + K] : (_Float16)0.f;
  }
  if (t < 64) {
    int K = t, off = 0;
    if (K < 45) {
      int c = K / 9, tap = K % 9;
      off = ((tap/3)*5 + c)*258 + (tap % 3);                    // +d gives padded col
    }
    soff[K] = off;
  }
  __syncthreads();

  // ---- phase 2: build im2col in f16 (branch-free: cndmask only)
  for (int f = t; f < 256*64; f += 512) {
    int d = f >> 6, K = f & 63;
    float x = xtp[soff[K] + d];
    float v = (K < 45) ? x : ((K == 45) ? 1.0f : 0.f);
    icol[f] = (_Float16)v;
  }
  __syncthreads();

  // ---- phase 3: WMMA. A operands: two contiguous 8-half chunks per lane.
  v16h a1[2][2];
  #pragma unroll
  for (int mt = 0; mt < 2; ++mt) {
    const _Float16* pm = w1p + (mt*16 + lm)*64;
    #pragma unroll
    for (int kt = 0; kt < 2; ++kt) {
      v8h lo = *(const v8h*)(pm + kt*32 + lh*16);
      v8h hi = *(const v8h*)(pm + kt*32 + lh*16 + 8);
      a1[mt][kt] = __builtin_shufflevector(lo, hi,
                     0,1,2,3,4,5,6,7,8,9,10,11,12,13,14,15);
    }
  }
  v8h a2lo = *(const v8h*)(w2p + lm*32 + lh*16);
  v8h a2hi = *(const v8h*)(w2p + lm*32 + lh*16 + 8);
  v16h a2 = __builtin_shufflevector(a2lo, a2hi,
              0,1,2,3,4,5,6,7,8,9,10,11,12,13,14,15);

  // B operands: lane holds 16 contiguous K halves of its pixel column
  int d0 = wv * 16, n = lm, d = d0 + n;
  const _Float16* bp = icol + d*64;
  v16h bb0 = *(const v16h*)(bp + lh*16);        // K 0..31 block
  v16h bb1 = *(const v16h*)(bp + 32 + lh*16);   // K 32..63 block

  v8f acc0 = {}, acc1 = {};
  acc0 = __builtin_amdgcn_wmma_f32_16x16x32_f16(false, a1[0][0], false, bb0, (short)0, acc0, false, false);
  acc1 = __builtin_amdgcn_wmma_f32_16x16x32_f16(false, a1[1][0], false, bb0, (short)0, acc1, false, false);
  acc0 = __builtin_amdgcn_wmma_f32_16x16x32_f16(false, a1[0][1], false, bb1, (short)0, acc0, false, false);
  acc1 = __builtin_amdgcn_wmma_f32_16x16x32_f16(false, a1[1][1], false, bb1, (short)0, acc1, false, false);

  // relu (bias already in acc), pack f16, store transposed [n][m] into this
  // wave's OWN exhausted im2col slice (exclusive; same-wave DS is in-order)
  _Float16* hb = icol + wv*1024;
  v8h ph0, ph1;
  #pragma unroll
  for (int v = 0; v < 8; ++v) {
    ph0[v] = (_Float16)fmaxf(acc0[v], 0.f);     // m = lh*8 + v
    ph1[v] = (_Float16)fmaxf(acc1[v], 0.f);     // m = 16 + lh*8 + v
  }
  *(v8h*)(hb + n*32 + lh*8)      = ph0;
  *(v8h*)(hb + n*32 + 16 + lh*8) = ph1;

  v16h b2v = *(const v16h*)(hb + n*32 + lh*16); // K(=h ch) lh*16..+15, pixel n
  v8f accD = {};
  accD = __builtin_amdgcn_wmma_f32_16x16x32_f16(false, a2, false, b2v, (short)0, accD, false, false);

  // x_new = x + dx + b2 ; D rows 0..4 live in VGPRs 0..4 of lanes 0..15
  if (lane < 16) {
    float* xo = xout + (size_t)b*5*JD;
    #pragma unroll
    for (int o = 0; o < 5; ++o)
      xo[(size_t)o*JD + j*256 + d] = xtp[(5 + o)*258 + d + 1] + accD[o] + b2[o];
  }
}

// ---------------------------------------------------------------- epilog ---
__global__ void k_colmean(const float* __restrict__ x, float* __restrict__ cm) {
  __shared__ float red[256];
  int bc = blockIdx.x / JJ;
  int j  = blockIdx.x % JJ;
  int t  = threadIdx.x;
  red[t] = tanhf(x[(size_t)bc*JD + j*256 + t]);
  __syncthreads();
  for (int s = 128; s > 0; s >>= 1) {
    if (t < s) red[t] += red[t + s];
    __syncthreads();
  }
  if (t == 0) cm[bc*JJ + j] = red[0] * (1.f/256.f);
}

__global__ void k_pool(const float* __restrict__ cm, float* __restrict__ out) {
  int t = blockIdx.x * 256 + threadIdx.x;
  if (t >= 5*BATCH*81) return;
  int c = t / (BATCH*81);
  int rem = t % (BATCH*81);
  int b = rem / 81, o = rem % 81;
  int s = (o * JJ) / 81;
  int e = ((o + 1) * JJ + 80) / 81;
  float acc = 0.f;
  for (int j = s; j < e; ++j) acc += cm[(b*5 + c)*JJ + j];
  out[t] = acc / (float)(e - s);
}

// ------------------------------------------------------------------ launch --
extern "C" void kernel_launch(void* const* d_in, const int* in_sizes, int n_in,
                              void* d_out, int out_size, void* d_ws, size_t ws_size,
                              hipStream_t stream) {
  const float* din  = (const float*)d_in[0];   // [8,36,9]
  const float* sin_ = (const float*)d_in[1];   // [8,9,9]
  const float* mi   = (const float*)d_in[2];   // [8]
  const float* mo   = (const float*)d_in[3];   // [8]
  const float* rbf  = (const float*)d_in[4];   // [64,421,256]
  const float* hdc  = (const float*)d_in[5];   // [8,421,256]
  const float* cw   = (const float*)d_in[6];   // [5,64]
  const float* cb   = (const float*)d_in[7];   // [5]
  const float* w1   = (const float*)d_in[8];   // [32,5,3,3] == [32][45]
  const float* b1   = (const float*)d_in[9];   // [32]
  const float* w2   = (const float*)d_in[10];  // [5,32]
  const float* b2   = (const float*)d_in[11];  // [5]

  float* ws = (float*)d_ws;
  float* S  = ws;                                   // 16 floats (8 scalars + 5 C)
  float* X0 = ws + 16;                              // [8,5,421,256]
  float* X1 = X0 + (size_t)BATCH*5*JD;              // ping-pong
  float* CM = X1 + (size_t)BATCH*5*JD;              // [8,5,421]

  k_scalars<<<1, 64, 0, stream>>>(din, sin_, mi, mo, cw, S);
  k_init<<<JD/256, 256, 0, stream>>>(rbf, hdc, cw, cb, S, X0);
  for (int step = 0; step < NSTEPS; ++step) {
    const float* xi = (step & 1) ? X1 : X0;
    float*       xo = (step & 1) ? X0 : X1;
    k_nca<<<BATCH*JJ, 512, 0, stream>>>(xi, xo, w1, b1, w2, b2);
  }
  // NSTEPS even -> final state back in X0
  k_colmean<<<BATCH*5*JJ, 256, 0, stream>>>(X0, CM);
  k_pool<<<(5*BATCH*81 + 255)/256, 256, 0, stream>>>(CM, (float*)d_out);
}
